// Attention_73254962201206
// MI455X (gfx1250) — matrix operand order repaired and checked by reading, not compiled, not run on hardware
//
#include <hip/hip_runtime.h>
#include <hip/hip_bf16.h>
#include <math.h>

#define BB   4
#define NN   2048
#define HH   16
#define DHH  64
#define CC   1024
#define MM   (BB * NN)   // 8192

typedef __attribute__((ext_vector_type(16))) __bf16 bf16x16;
typedef __attribute__((ext_vector_type(8)))  __bf16 bf16x8;
typedef __attribute__((ext_vector_type(2)))  __bf16 bf16x2;
typedef __attribute__((ext_vector_type(8)))  float  f32x8;

__device__ __forceinline__ f32x8 wmma_bf16(bf16x16 a, bf16x16 b, f32x8 c) {
  return __builtin_amdgcn_wmma_f32_16x16x32_bf16(false, a, false, b, (short)0, c, false, false);
}

__device__ __forceinline__ bf16x16 cat8(bf16x8 a, bf16x8 b) {
  return __builtin_shufflevector(a, b, 0, 1, 2, 3, 4, 5, 6, 7, 8, 9, 10, 11, 12, 13, 14, 15);
}

// A fragment (16x32 bf16): per ISA table, lane holds K = 8*hi+0..7 and
// K = 16+8*hi+0..7 -> two contiguous 16B runs in a row-major [row][k] tile.
// Requires stride % 8 == 0 for ds_load_b128 alignment.
__device__ __forceinline__ bf16x16 load_a_frag(const __bf16* base, int stride, int row, int lane) {
  const __bf16* p = base + (row + (lane & 15)) * stride + 8 * (lane >> 4);
  bf16x8 lo = *(const bf16x8*)(p);
  bf16x8 hi = *(const bf16x8*)(p + 16);
  return cat8(lo, hi);
}

// B fragment (32x16 bf16): lane holds K = 8*hi+0..7 and K = 16+8*hi+0..7, n fixed
// -> two contiguous 16B runs when the tile is stored TRANSPOSED as [n][k].
__device__ __forceinline__ bf16x16 load_b_frag_t(const __bf16* base, int stride, int col, int lane) {
  const __bf16* p = base + (col + (lane & 15)) * stride + 8 * (lane >> 4);
  bf16x8 lo = *(const bf16x8*)(p);
  bf16x8 hi = *(const bf16x8*)(p + 16);
  return cat8(lo, hi);
}

// ---------------------------------------------------------------------------
// Kernel 1: fused QKV projection + RoPE + bf16 pack.
// [8192 x 1024] x [1024 x 3072] (Wq | Wkv), 128x64 tile/block, K-step 64.
// ---------------------------------------------------------------------------
#define T1M 128
#define T1N 64
#define T1K 64
#define AST 72
#define BST 72

__global__ __launch_bounds__(256) void qkv_rope_kernel(
    const float* __restrict__ x, const float* __restrict__ Wq, const float* __restrict__ Wkv,
    __bf16* __restrict__ Qb, __bf16* __restrict__ Kb, __bf16* __restrict__ Vb) {
  __shared__ alignas(16) __bf16 As[T1M][AST];   // row-major [m][k]
  __shared__ alignas(16) __bf16 Bst[T1N][BST];  // transposed [n][k]
  __shared__ float Yt[T1M][T1N];

  const int tid = threadIdx.x, lane = tid & 31, wave = tid >> 5, hi = lane >> 4;
  const int m0 = blockIdx.y * T1M;
  const int j0 = blockIdx.x * T1N;              // global col in [0, 3072)
  const int wm = (wave >> 1) * 32, wn = (wave & 1) * 32;

  f32x8 acc[2][2];
#pragma unroll
  for (int i = 0; i < 2; ++i)
#pragma unroll
    for (int j = 0; j < 2; ++j)
#pragma unroll
      for (int e = 0; e < 8; ++e) acc[i][j][e] = 0.0f;

  // Whole block lives in one weight section (tile is 64-aligned, sections 1024-wide).
  const float* wbase = (j0 < CC) ? (Wq + j0) : (Wkv + (j0 - CC));
  const int    wld   = (j0 < CC) ? CC : (2 * CC);

  for (int kk = 0; kk < CC; kk += T1K) {
    // A: 128x64 f32 -> bf16, packed pair stores
    for (int i = tid; i < T1M * T1K / 2; i += 256) {
      int r = i >> 5, c = (i & 31) * 2;
      float2 v = *(const float2*)&x[(size_t)(m0 + r) * CC + kk + c];
      bf16x2 pk; pk[0] = (__bf16)v.x; pk[1] = (__bf16)v.y;
      *(bf16x2*)&As[r][c] = pk;
    }
    // B: 64x64, stored transposed
    for (int i = tid; i < T1K * T1N / 2; i += 256) {
      int r = i >> 5, c = (i & 31) * 2;          // r = k row, c = first n col
      float2 w = *(const float2*)&wbase[(size_t)(kk + r) * wld + c];
      Bst[c][r]     = (__bf16)w.x;
      Bst[c + 1][r] = (__bf16)w.y;
    }
    if (kk + T1K < CC) {
      __builtin_prefetch(&x[(size_t)(m0 + (tid >> 1)) * CC + kk + T1K], 0, 1);
    }
    __syncthreads();

#pragma unroll
    for (int kc = 0; kc < T1K; kc += 32) {
      bf16x16 bf0 = load_b_frag_t(&Bst[0][kc], BST, wn, lane);
      bf16x16 bf1 = load_b_frag_t(&Bst[0][kc], BST, wn + 16, lane);
#pragma unroll
      for (int ti = 0; ti < 2; ++ti) {
        bf16x16 af = load_a_frag(&As[0][kc], AST, wm + ti * 16, lane);
        acc[ti][0] = wmma_bf16(af, bf0, acc[ti][0]);
        acc[ti][1] = wmma_bf16(af, bf1, acc[ti][1]);
      }
    }
    __syncthreads();
  }

  // Spill so RoPE can pair columns d <-> d +/- 32 across waves.
#pragma unroll
  for (int ti = 0; ti < 2; ++ti)
#pragma unroll
    for (int tj = 0; tj < 2; ++tj)
#pragma unroll
      for (int e = 0; e < 8; ++e)
        Yt[wm + ti * 16 + 8 * hi + e][wn + tj * 16 + (lane & 15)] = acc[ti][tj][e];
  __syncthreads();

  const int sect = j0 >> 10;                 // 0 = q, 1 = k, 2 = v
  const int h    = (j0 >> 6) & (HH - 1);     // tile width 64 == one head
  for (int i = tid; i < T1M * T1N; i += 256) {
    int r = i >> 6, d = i & 63;
    int m = m0 + r;
    int b = m >> 11, n = m & (NN - 1);
    float  y    = Yt[r][d];
    size_t oidx = (((size_t)(b * HH + h)) * NN + n) * DHH + d;
    if (sect == 2) {
      Vb[oidx] = (__bf16)y;
    } else {
      int   p  = (d < 32) ? d + 32 : d - 32;
      float yp = Yt[r][p];
      float inv_freq = __powf(10000.0f, -(float)(2 * (d & 31)) * (1.0f / 64.0f));
      float sn, cs;
      __sincosf((float)n * inv_freq, &sn, &cs);
      float rot = (d < 32) ? -yp : yp;
      float v   = y * cs + rot * sn;
      if (sect == 0) Qb[oidx] = (__bf16)(v * 0.125f);   // fold 1/sqrt(DH)
      else           Kb[oidx] = (__bf16)v;
    }
  }
}

// ---------------------------------------------------------------------------
// Kernel 2: flash attention. 8 waves; wave owns 16 q-rows; KTILE = 64 keys.
// ---------------------------------------------------------------------------
#define QT  128
#define KT2 64
#define QST 72
#define KST 72
#define VST 72
#define PST 72

__global__ __launch_bounds__(256) void flash_attn_kernel(
    const __bf16* __restrict__ Qb, const __bf16* __restrict__ Kb,
    const __bf16* __restrict__ Vb, __bf16* __restrict__ Ob) {
  __shared__ alignas(16) __bf16 Qs[QT][QST];    // [q][d]
  __shared__ alignas(16) __bf16 Ks[KT2][KST];   // [key][d]  (== transposed for K^T B-frags)
  __shared__ alignas(16) __bf16 Vt[DHH][VST];   // [d][key]  (transposed for V B-frags)
  __shared__ alignas(16) __bf16 Ps[8][16][PST]; // per-wave C-layout -> A-layout bounce

  const int tid = threadIdx.x, lane = tid & 31, wave = tid >> 5, hi = lane >> 4;
  const int bh = blockIdx.y;
  const int q0 = blockIdx.x * QT;
  const size_t head = (size_t)bh * NN * DHH;
  const __bf16* Qh = Qb + head;
  const __bf16* Kh = Kb + head;
  const __bf16* Vh = Vb + head;

  for (int i = tid; i < QT * DHH / 8; i += 256) {
    int r = i >> 3, c = (i & 7) * 8;
    *(bf16x8*)&Qs[r][c] = *(const bf16x8*)&Qh[(size_t)(q0 + r) * DHH + c];
  }
  __syncthreads();

  bf16x16 qf0 = load_a_frag(&Qs[0][0],  QST, wave * 16, lane);
  bf16x16 qf1 = load_a_frag(&Qs[0][32], QST, wave * 16, lane);

  float mrow[8], lrow[8];
  f32x8 o[4];
#pragma unroll
  for (int e = 0; e < 8; ++e) { mrow[e] = -3.0e38f; lrow[e] = 0.0f; }
#pragma unroll
  for (int t = 0; t < 4; ++t)
#pragma unroll
    for (int e = 0; e < 8; ++e) o[t][e] = 0.0f;

  for (int k0 = 0; k0 < NN; k0 += KT2) {
    __syncthreads();  // protect Ks/Vt reuse across iterations
    for (int i = tid; i < KT2 * DHH / 8; i += 256) {
      int r = i >> 3, c = (i & 7) * 8;
      *(bf16x8*)&Ks[r][c] = *(const bf16x8*)&Kh[(size_t)(k0 + r) * DHH + c];
      bf16x8 v = *(const bf16x8*)&Vh[(size_t)(k0 + r) * DHH + c];
#pragma unroll
      for (int u = 0; u < 8; ++u) Vt[c + u][r] = v[u];
    }
    __syncthreads();

    // S = Q * K^T : four 16x16 tiles over 64 keys, contraction over DH = 64.
    f32x8 s[4];
#pragma unroll
    for (int t = 0; t < 4; ++t)
#pragma unroll
      for (int e = 0; e < 8; ++e) s[t][e] = 0.0f;
#pragma unroll
    for (int t = 0; t < 4; ++t) {
      bf16x16 kb0 = load_b_frag_t(&Ks[0][0],  KST, t * 16, lane);
      bf16x16 kb1 = load_b_frag_t(&Ks[0][32], KST, t * 16, lane);
      s[t] = wmma_bf16(qf0, kb0, s[t]);
      s[t] = wmma_bf16(qf1, kb1, s[t]);
    }

    // Online softmax: row e lives across the 16 lanes of this half-wave.
#pragma unroll
    for (int e = 0; e < 8; ++e) {
      float mx = fmaxf(fmaxf(s[0][e], s[1][e]), fmaxf(s[2][e], s[3][e]));
#pragma unroll
      for (int off = 1; off < 16; off <<= 1) mx = fmaxf(mx, __shfl_xor(mx, off, 32));
      float mn   = fmaxf(mrow[e], mx);
      float corr = __expf(mrow[e] - mn);
      float p0 = __expf(s[0][e] - mn);
      float p1 = __expf(s[1][e] - mn);
      float p2 = __expf(s[2][e] - mn);
      float p3 = __expf(s[3][e] - mn);
      s[0][e] = p0; s[1][e] = p1; s[2][e] = p2; s[3][e] = p3;
      float rs = (p0 + p1) + (p2 + p3);
#pragma unroll
      for (int off = 1; off < 16; off <<= 1) rs += __shfl_xor(rs, off, 32);
      lrow[e] = lrow[e] * corr + rs;
      mrow[e] = mn;
      o[0][e] *= corr; o[1][e] *= corr; o[2][e] *= corr; o[3][e] *= corr;
    }

    // Re-shape P: C-layout -> A-layout through per-wave LDS (in-order LDS).
#pragma unroll
    for (int t = 0; t < 4; ++t)
#pragma unroll
      for (int e = 0; e < 8; ++e)
        Ps[wave][8 * hi + e][t * 16 + (lane & 15)] = (__bf16)s[t][e];

    bf16x16 pa0 = load_a_frag(&Ps[wave][0][0],  PST, 0, lane);
    bf16x16 pa1 = load_a_frag(&Ps[wave][0][32], PST, 0, lane);
#pragma unroll
    for (int t = 0; t < 4; ++t) {
      bf16x16 vb0 = load_b_frag_t(&Vt[0][0],  VST, t * 16, lane);
      bf16x16 vb1 = load_b_frag_t(&Vt[0][32], VST, t * 16, lane);
      o[t] = wmma_bf16(pa0, vb0, o[t]);
      o[t] = wmma_bf16(pa1, vb1, o[t]);
    }
  }

  // Normalize and scatter to [B*N, C] bf16 (undoes the head transpose).
  const int b = bh >> 4, h = bh & 15;
#pragma unroll
  for (int e = 0; e < 8; ++e) {
    float invl = 1.0f / lrow[e];
    int qrow = q0 + wave * 16 + 8 * hi + e;
#pragma unroll
    for (int t = 0; t < 4; ++t) {
      int d = t * 16 + (lane & 15);
      Ob[((size_t)(b * NN + qrow)) * CC + h * DHH + d] = (__bf16)(o[t][e] * invl);
    }
  }
}

// ---------------------------------------------------------------------------
// Kernel 3: out projection  out = O(bf16) @ Wout + b_out  -> f32
// ---------------------------------------------------------------------------
__global__ __launch_bounds__(256) void out_proj_kernel(
    const __bf16* __restrict__ Ob, const float* __restrict__ Wout,
    const float* __restrict__ bout, float* __restrict__ out) {
  __shared__ alignas(16) __bf16 As[T1M][AST];
  __shared__ alignas(16) __bf16 Bst[T1N][BST];

  const int tid = threadIdx.x, lane = tid & 31, wave = tid >> 5, hi = lane >> 4;
  const int m0 = blockIdx.y * T1M;
  const int j0 = blockIdx.x * T1N;
  const int wm = (wave >> 1) * 32, wn = (wave & 1) * 32;

  f32x8 acc[2][2];
#pragma unroll
  for (int i = 0; i < 2; ++i)
#pragma unroll
    for (int j = 0; j < 2; ++j)
#pragma unroll
      for (int e = 0; e < 8; ++e) acc[i][j][e] = 0.0f;

  for (int kk = 0; kk < CC; kk += T1K) {
    for (int i = tid; i < T1M * T1K / 8; i += 256) {
      int r = i >> 3, c = (i & 7) * 8;
      *(bf16x8*)&As[r][c] = *(const bf16x8*)&Ob[(size_t)(m0 + r) * CC + kk + c];
    }
    for (int i = tid; i < T1K * T1N / 2; i += 256) {
      int r = i >> 5, c = (i & 31) * 2;
      float2 w = *(const float2*)&Wout[(size_t)(kk + r) * CC + j0 + c];
      Bst[c][r]     = (__bf16)w.x;
      Bst[c + 1][r] = (__bf16)w.y;
    }
    if (kk + T1K < CC) {
      __builtin_prefetch(&Wout[(size_t)(kk + T1K + (tid >> 2)) * CC + j0], 0, 1);
    }
    __syncthreads();

#pragma unroll
    for (int kc = 0; kc < T1K; kc += 32) {
      bf16x16 bf0 = load_b_frag_t(&Bst[0][kc], BST, wn, lane);
      bf16x16 bf1 = load_b_frag_t(&Bst[0][kc], BST, wn + 16, lane);
#pragma unroll
      for (int ti = 0; ti < 2; ++ti) {
        bf16x16 af = load_a_frag(&As[0][kc], AST, wm + ti * 16, lane);
        acc[ti][0] = wmma_bf16(af, bf0, acc[ti][0]);
        acc[ti][1] = wmma_bf16(af, bf1, acc[ti][1]);
      }
    }
    __syncthreads();
  }

#pragma unroll
  for (int ti = 0; ti < 2; ++ti)
#pragma unroll
    for (int tj = 0; tj < 2; ++tj)
#pragma unroll
      for (int e = 0; e < 8; ++e) {
        int row = m0 + wm + ti * 16 + 8 * hi + e;
        int col = j0 + wn + tj * 16 + (lane & 15);
        out[(size_t)row * CC + col] = acc[ti][tj][e] + bout[col];
      }
}

// ---------------------------------------------------------------------------
extern "C" void kernel_launch(void* const* d_in, const int* in_sizes, int n_in,
                              void* d_out, int out_size, void* d_ws, size_t ws_size,
                              hipStream_t stream) {
  const float* x    = (const float*)d_in[0];
  const float* Wq   = (const float*)d_in[1];
  const float* Wkv  = (const float*)d_in[2];
  const float* Wout = (const float*)d_in[3];
  const float* bout = (const float*)d_in[4];
  float* out = (float*)d_out;

  char* ws = (char*)d_ws;
  const size_t elems = (size_t)MM * CC;      // 8M elements per tensor
  __bf16* Qb = (__bf16*)(ws);
  __bf16* Kb = (__bf16*)(ws + elems * 2);
  __bf16* Vb = (__bf16*)(ws + elems * 4);
  __bf16* Ob = (__bf16*)(ws + elems * 6);    // 64 MB total bf16 scratch

  qkv_rope_kernel<<<dim3((3 * CC) / T1N, MM / T1M), 256, 0, stream>>>(x, Wq, Wkv, Qb, Kb, Vb);
  flash_attn_kernel<<<dim3(NN / QT, BB * HH), 256, 0, stream>>>(Qb, Kb, Vb, Ob);
  out_proj_kernel<<<dim3(CC / T1N, MM / T1M), 256, 0, stream>>>(Ob, Wout, bout, out);

  (void)in_sizes; (void)n_in; (void)out_size; (void)ws_size;
}
